// RoutingAttentionModel_7430293422543
// MI455X (gfx1250) — matrix operand order repaired and checked by reading, not compiled, not run on hardware
//
#include <hip/hip_runtime.h>
#include <hip/hip_bf16.h>
#include <math.h>

// ---------------- model dims ----------------
#define BSZ 256
#define NN  256
#define DD  128
#define HH  8
#define LL  3
#define FFD 512
#define TOK (BSZ * NN)   // 65536

typedef __attribute__((ext_vector_type(16))) __bf16 v16bf;
typedef __attribute__((ext_vector_type(8)))  float  v8f;
typedef int v4i __attribute__((vector_size(16)));

union FragBF { v16bf v; uint4 q[2]; };
static_assert(sizeof(FragBF) == 32, "frag size");

#if __has_builtin(__builtin_amdgcn_global_load_async_to_lds_b128) && \
    __has_builtin(__builtin_amdgcn_s_wait_asynccnt)
#define USE_ASYNC_LDS 1
typedef __attribute__((address_space(1))) v4i* gv4i_t;
typedef __attribute__((address_space(3))) v4i* lv4i_t;
#endif

// =====================================================================
// f32 -> bf16 weight conversion
// =====================================================================
__global__ void f32_to_bf16_kernel(const float* __restrict__ src,
                                   __bf16* __restrict__ dst, int n) {
  int i = blockIdx.x * 256 + threadIdx.x;
  if (i < n) dst[i] = (__bf16)src[i];
}

// =====================================================================
// input embedding: x[tok,d] = c0*W[d,0] + c1*W[d,1] + b[d]
// =====================================================================
__global__ void embed_kernel(const float* __restrict__ coords,
                             const float* __restrict__ W,
                             const float* __restrict__ bias,
                             float* __restrict__ x) {
  int i = blockIdx.x * 256 + threadIdx.x;       // over TOK*DD
  int tok = i >> 7, d = i & 127;
  float c0 = coords[tok * 2 + 0];
  float c1 = coords[tok * 2 + 1];
  x[i] = c0 * W[d * 2 + 0] + c1 * W[d * 2 + 1] + bias[d];
}

// =====================================================================
// GEMM: C[M,N] = A[M,K](f32) * W[N,K]^T(bf16) + bias, optional exact gelu
// WG = 256 thr = 8 waves; tile 128 rows x NT*16 cols; K-step 32.
// Each wave: one 16-row strip, NT independent WMMA accumulators.
// =====================================================================
template <int NT>
__global__ __launch_bounds__(256)
void gemm_bf16_kernel(const float* __restrict__ A,
                      const __bf16* __restrict__ W,
                      const float* __restrict__ bias,
                      float* __restrict__ C,
                      int N, int K, int act) {
  __shared__ alignas(16) __bf16 Asm[128 * 40];       // row stride 40 elems (80B)
  __shared__ alignas(16) __bf16 Bsm[NT * 16 * 40];

  const int tid  = threadIdx.x;
  const int w    = tid >> 5;
  const int lane = tid & 31;
  const int r    = lane & 15;
  const int hi   = lane >> 4;
  const int rowBase = blockIdx.y * 128;
  const int colBase = blockIdx.x * (NT * 16);

  v8f acc[NT] = {};

  for (int k0 = 0; k0 < K; k0 += 32) {
    // stage A tile (convert f32 -> bf16)
    {
      int arow = tid >> 1, cg = (tid & 1) * 16;
      const float* src = A + (size_t)(rowBase + arow) * K + k0 + cg;
      __bf16* dst = Asm + arow * 40 + cg;
#pragma unroll
      for (int j = 0; j < 16; ++j) dst[j] = (__bf16)src[j];
      if (k0 + 32 < K) __builtin_prefetch(src + 32, 0, 1);  // global_prefetch_b8
    }
    // stage W tile (already bf16): async DMA to LDS when available
#ifdef USE_ASYNC_LDS
    for (int i = tid; i < NT * 64; i += 256) {
      int brow = i >> 2, cg = (i & 3) * 8;                  // 8 bf16 = 16B chunks
      const __bf16* src = W + (size_t)(colBase + brow) * K + k0 + cg;
      __bf16* dst = Bsm + brow * 40 + cg;
      __builtin_amdgcn_global_load_async_to_lds_b128((gv4i_t)(v4i*)src,
                                                     (lv4i_t)(v4i*)dst, 0, 0);
    }
    __builtin_amdgcn_s_wait_asynccnt(0);
#else
    for (int i = tid; i < NT * 16 * 8; i += 256) {
      int brow = i >> 3, cg = (i & 7) * 4;
      const __bf16* src = W + (size_t)(colBase + brow) * K + k0 + cg;
      __bf16* dst = Bsm + brow * 40 + cg;
#pragma unroll
      for (int j = 0; j < 4; ++j) dst[j] = src[j];
    }
#endif
    __syncthreads();

    FragBF af;
    const char* ab = (const char*)Asm;
    const char* bb = (const char*)Bsm;
    int arow = w * 16 + r;
    // A layout: lane half hi -> K 16*hi .. 16*hi+15
    af.q[0] = *(const uint4*)(ab + arow * 80 + 32 * hi);
    af.q[1] = *(const uint4*)(ab + arow * 80 + 32 * hi + 16);
#pragma unroll
    for (int nt = 0; nt < NT; ++nt) {
      // B layout: lane = column, K = 16*hi .. 16*hi+15
      FragBF bf;
      bf.q[0] = *(const uint4*)(bb + (nt * 16 + r) * 80 + 32 * hi);
      bf.q[1] = *(const uint4*)(bb + (nt * 16 + r) * 80 + 32 * hi + 16);
      acc[nt] = __builtin_amdgcn_wmma_f32_16x16x32_bf16(false, af.v, false, bf.v,
                                                        (short)0, acc[nt], false, false);
    }
    __syncthreads();
  }

#pragma unroll
  for (int nt = 0; nt < NT; ++nt) {
    int col = colBase + nt * 16 + r;
    float bv = bias ? bias[col] : 0.0f;
#pragma unroll
    for (int i = 0; i < 8; ++i) {
      int row = rowBase + w * 16 + i + 8 * hi;   // C layout: M = i + 8*hi
      float v = acc[nt][i] + bv;
      if (act == 1) v = 0.5f * v * (1.0f + erff(v * 0.70710678118654752f));
      C[(size_t)row * N + col] = v;
    }
  }
}

// =====================================================================
// attention: one WG per (b,h). dh=16 zero-padded to K=32 for scores.
// =====================================================================
#define ATTN_LDS (20480 + 8704 + 8 * 8704)   // ksm + vt + per-wave attn

__global__ __launch_bounds__(256)
void attention_kernel(const float* __restrict__ qkv, float* __restrict__ o) {
  extern __shared__ char smem_dyn[];
  __bf16* ksm     = (__bf16*)smem_dyn;                    // 256 x stride40
  __bf16* vt      = (__bf16*)(smem_dyn + 20480);          // 16  x stride272
  __bf16* attnAll = (__bf16*)(smem_dyn + 20480 + 8704);   // 8 * (16 x 272)

  const int tid  = threadIdx.x;
  const int b    = blockIdx.x >> 3;
  const int h    = blockIdx.x & 7;
  const int wave = tid >> 5, lane = tid & 31;
  const int r = lane & 15, hi = lane >> 4;
  const size_t base = (size_t)b * NN * (3 * DD);

  // stage K head (bf16, K padded 16->32 with zeros)
  {
    const float* src = qkv + base + (size_t)tid * (3 * DD) + DD + h * 16;
    __bf16* dst = ksm + tid * 40;
#pragma unroll
    for (int d = 0; d < 16; ++d) dst[d] = (__bf16)src[d];
#pragma unroll
    for (int d = 16; d < 32; ++d) dst[d] = (__bf16)0.0f;
  }
  // stage V^T head
  {
    int d = tid >> 4, jb = (tid & 15) * 16;
#pragma unroll
    for (int j = 0; j < 16; ++j)
      vt[d * 272 + jb + j] =
          (__bf16)qkv[base + (size_t)(jb + j) * (3 * DD) + 2 * DD + h * 16 + d];
  }
  __syncthreads();

  __bf16* attnsm = attnAll + wave * (16 * 272);

  for (int rt = 0; rt < 2; ++rt) {
    const int rowTile = wave * 2 + rt;
    const int grow = rowTile * 16 + r;

    // q fragment straight from global (A layout), K>=16 zero
    FragBF qf;
#pragma unroll
    for (int e = 0; e < 16; ++e) {
      int kk = 16 * hi + e;
      qf.v[e] = (kk < 16)
                    ? (__bf16)qkv[base + (size_t)grow * (3 * DD) + h * 16 + kk]
                    : (__bf16)0.0f;
    }

    v8f ct[16] = {};
    const char* kb = (const char*)ksm;
#pragma unroll
    for (int j = 0; j < 16; ++j) {
      FragBF kf;
      int n = j * 16 + r;
      kf.q[0] = *(const uint4*)(kb + n * 80 + 32 * hi);
      kf.q[1] = *(const uint4*)(kb + n * 80 + 32 * hi + 16);
      ct[j] = __builtin_amdgcn_wmma_f32_16x16x32_bf16(false, qf.v, false, kf.v,
                                                      (short)0, ct[j], false, false);
    }

    // softmax over 256 cols of each of 8 rows this lane holds
#pragma unroll
    for (int i = 0; i < 8; ++i) {
      float m = -3.0e38f;
#pragma unroll
      for (int j = 0; j < 16; ++j) m = fmaxf(m, ct[j][i]);
#pragma unroll
      for (int s = 1; s < 16; s <<= 1) m = fmaxf(m, __shfl_xor(m, s, 32));
      float sum = 0.0f;
#pragma unroll
      for (int j = 0; j < 16; ++j) {
        float p = __expf((ct[j][i] - m) * 0.25f);  // 1/sqrt(16)
        ct[j][i] = p; sum += p;
      }
#pragma unroll
      for (int s = 1; s < 16; s <<= 1) sum += __shfl_xor(sum, s, 32);
      float inv = 1.0f / sum;
#pragma unroll
      for (int j = 0; j < 16; ++j)
        attnsm[(i + 8 * hi) * 272 + j * 16 + r] = (__bf16)(ct[j][i] * inv);
    }
    __syncthreads();

    // O = P(16x256) * V(256x16)
    v8f oacc = {};
    const char* abuf = (const char*)attnsm;
    const char* vbuf = (const char*)vt;
#pragma unroll
    for (int kk = 0; kk < 8; ++kk) {
      FragBF afr, vfr;
      afr.q[0] = *(const uint4*)(abuf + r * 544 + kk * 64 + 32 * hi);
      afr.q[1] = *(const uint4*)(abuf + r * 544 + kk * 64 + 32 * hi + 16);
      vfr.q[0] = *(const uint4*)(vbuf + r * 544 + kk * 64 + 32 * hi);
      vfr.q[1] = *(const uint4*)(vbuf + r * 544 + kk * 64 + 32 * hi + 16);
      oacc = __builtin_amdgcn_wmma_f32_16x16x32_bf16(false, afr.v, false, vfr.v,
                                                     (short)0, oacc, false, false);
    }
#pragma unroll
    for (int i = 0; i < 8; ++i) {
      int row = rowTile * 16 + i + 8 * hi;
      o[((size_t)b * NN + row) * DD + h * 16 + r] = oacc[i];
    }
    __syncthreads();
  }
}

// =====================================================================
// y = LayerNorm(x + delta) * g + b   (one block per token row, D=128)
// =====================================================================
__global__ __launch_bounds__(128)
void ln_kernel(const float* __restrict__ x, const float* __restrict__ dlt,
               const float* __restrict__ g, const float* __restrict__ bta,
               float* __restrict__ out) {
  __shared__ float red[128];
  int row = blockIdx.x, t = threadIdx.x;
  size_t idx = (size_t)row * DD + t;
  float v = x[idx] + dlt[idx];
  red[t] = v; __syncthreads();
  for (int off = 64; off > 0; off >>= 1) {
    if (t < off) red[t] += red[t + off];
    __syncthreads();
  }
  float mu = red[0] * (1.0f / 128.0f);
  __syncthreads();
  float c = v - mu;
  red[t] = c * c; __syncthreads();
  for (int off = 64; off > 0; off >>= 1) {
    if (t < off) red[t] += red[t + off];
    __syncthreads();
  }
  float var = red[0] * (1.0f / 128.0f);
  out[idx] = c * rsqrtf(var + 1e-5f) * g[t] + bta[t];
}

// graph embedding: mean over nodes
__global__ __launch_bounds__(128)
void mean_kernel(const float* __restrict__ x, float* __restrict__ gemb) {
  int b = blockIdx.x, t = threadIdx.x;
  float s = 0.0f;
  for (int n = 0; n < NN; ++n) s += x[((size_t)b * NN + n) * DD + t];
  gemb[b * DD + t] = s * (1.0f / 256.0f);
}

// =====================================================================
// greedy decode: one WG per batch element, 256 sequential steps.
// keys (128KB f32) + Wq_prev (66KB f32) live in LDS for all 256 steps.
// =====================================================================
#define DEC_LDS ((256 * 129 + 128 * 129) * 4)

__global__ __launch_bounds__(256)
void decode_kernel(const float* __restrict__ node,
                   const float* __restrict__ keys,
                   const float* __restrict__ gemb,
                   const float* __restrict__ Wq,
                   float* __restrict__ out) {
  extern __shared__ char smem_dyn[];
  float* keysl = (float*)smem_dyn;          // 256 x stride 129
  float* wqp   = keysl + 256 * 129;         // 128 x stride 129

  __shared__ float logits[256], redv[256], query[128], pe[128], qg[128], gembS[128];
  __shared__ int   redi[256], visited[256], prevS;

  const int b = blockIdx.x, t = threadIdx.x;

  for (int i = t; i < 256 * 128; i += 256) {
    int n = i >> 7, d = i & 127;
    keysl[n * 129 + d] = keys[((size_t)b * NN + n) * DD + d];
  }
  for (int i = t; i < 128 * 128; i += 256) {
    int dd = i >> 7, j = i & 127;
    wqp[dd * 129 + j] = Wq[dd * 256 + 128 + j];
  }
  if (t < 128) gembS[t] = gemb[b * DD + t];
  visited[t] = 0;
  if (t == 0) prevS = 0;
  __syncthreads();
  if (t < 128) {                     // graph half of query, constant per batch
    float s = 0.0f;
    for (int j = 0; j < 128; ++j) s += Wq[t * 256 + j] * gembS[j];
    qg[t] = s;
  }
  __syncthreads();

  const float scale = 0.08838834764831845f;  // 1/sqrt(128)

  for (int step = 0; step < NN; ++step) {
    if (t < 128) pe[t] = node[((size_t)b * NN + prevS) * DD + t];
    __syncthreads();
    if (t < 128) {
      float s = qg[t];
      const float* wr = &wqp[t * 129];
      for (int j = 0; j < 128; ++j) s += wr[j] * pe[j];
      query[t] = s;
    }
    __syncthreads();
    {
      float s = 0.0f;
      const float* kr = &keysl[t * 129];
      for (int d = 0; d < 128; ++d) s += query[d] * kr[d];
      s *= scale;
      if (visited[t]) s = -1e30f;
      logits[t] = s; redv[t] = s; redi[t] = t;
    }
    __syncthreads();
    // argmax (strict > keeps the lower slot on ties, not always the first index)
    for (int off = 128; off > 0; off >>= 1) {
      if (t < off) {
        if (redv[t + off] > redv[t]) { redv[t] = redv[t + off]; redi[t] = redi[t + off]; }
      }
      __syncthreads();
    }
    float m = redv[0]; int a = redi[0];
    __syncthreads();
    redv[t] = expf(logits[t] - m);
    __syncthreads();
    for (int off = 128; off > 0; off >>= 1) {
      if (t < off) redv[t] += redv[t + off];
      __syncthreads();
    }
    if (t == 0) {
      float lp = logits[a] - m - logf(redv[0]);
      out[(size_t)b * NN + step] = (float)a;              // actions
      out[(size_t)TOK + (size_t)b * NN + step] = lp;      // log_probs
      visited[a] = 1; prevS = a;
    }
    __syncthreads();
  }
  if (t == 0) out[2 * (size_t)TOK + b] = 0.0f;            // entropy
}

// =====================================================================
// host orchestration
// =====================================================================
extern "C" void kernel_launch(void* const* d_in, const int* in_sizes, int n_in,
                              void* d_out, int out_size, void* d_ws, size_t ws_size,
                              hipStream_t stream) {
  (void)in_sizes; (void)n_in; (void)out_size; (void)ws_size;
  const float* coords = (const float*)d_in[0];
  const float* in_W   = (const float*)d_in[1];
  const float* in_b   = (const float*)d_in[2];
  const float* Wqkv   = (const float*)d_in[3];
  const float* bqkv   = (const float*)d_in[4];
  const float* Wo     = (const float*)d_in[5];
  const float* bo     = (const float*)d_in[6];
  const float* ln1g   = (const float*)d_in[7];
  const float* ln1b   = (const float*)d_in[8];
  const float* W1     = (const float*)d_in[9];
  const float* b1     = (const float*)d_in[10];
  const float* W2     = (const float*)d_in[11];
  const float* b2     = (const float*)d_in[12];
  const float* ln2g   = (const float*)d_in[13];
  const float* ln2b   = (const float*)d_in[14];
  const float* Wq     = (const float*)d_in[15];
  const float* Wk     = (const float*)d_in[16];
  float* out = (float*)d_out;

  char* ws = (char*)d_ws;
  size_t off = 0;
  auto alloc = [&](size_t bytes) -> void* {
    void* p = ws + off;
    off += (bytes + 255) & ~(size_t)255;
    return p;
  };
  float* x    = (float*)alloc((size_t)TOK * DD * 4);
  float* big  = (float*)alloc((size_t)TOK * FFD * 4);   // qkv (384) / ff1 (512)
  float* t1   = (float*)alloc((size_t)TOK * DD * 4);
  float* t2   = (float*)alloc((size_t)TOK * DD * 4);
  float* keys = (float*)alloc((size_t)TOK * DD * 4);
  float* gemb = (float*)alloc((size_t)BSZ * DD * 4);
  __bf16* wqkv_bf = (__bf16*)alloc((size_t)LL * 3 * DD * DD * 2);
  __bf16* wo_bf   = (__bf16*)alloc((size_t)LL * DD * DD * 2);
  __bf16* w1_bf   = (__bf16*)alloc((size_t)LL * FFD * DD * 2);
  __bf16* w2_bf   = (__bf16*)alloc((size_t)LL * DD * FFD * 2);
  __bf16* wk_bf   = (__bf16*)alloc((size_t)DD * DD * 2);

  auto cvt = [&](const float* s, __bf16* d, int n) {
    f32_to_bf16_kernel<<<(n + 255) / 256, 256, 0, stream>>>(s, d, n);
  };
  cvt(Wqkv, wqkv_bf, LL * 3 * DD * DD);
  cvt(Wo,   wo_bf,   LL * DD * DD);
  cvt(W1,   w1_bf,   LL * FFD * DD);
  cvt(W2,   w2_bf,   LL * DD * FFD);
  cvt(Wk,   wk_bf,   DD * DD);

  embed_kernel<<<(TOK * DD) / 256, 256, 0, stream>>>(coords, in_W, in_b, x);

  for (int l = 0; l < LL; ++l) {
    // QKV: [65536,128] x [384,128]^T ; strip 192 cols -> A read 2x
    gemm_bf16_kernel<12><<<dim3(2, TOK / 128), 256, 0, stream>>>(
        x, wqkv_bf + (size_t)l * 3 * DD * DD, bqkv + l * 3 * DD, big, 3 * DD, DD, 0);
    // attention
    attention_kernel<<<BSZ * HH, 256, ATTN_LDS, stream>>>(big, t1);
    // Wo projection (strip 128 -> A read once)
    gemm_bf16_kernel<8><<<dim3(1, TOK / 128), 256, 0, stream>>>(
        t1, wo_bf + (size_t)l * DD * DD, bo + l * DD, t2, DD, DD, 0);
    ln_kernel<<<TOK, 128, 0, stream>>>(x, t2, ln1g + l * DD, ln1b + l * DD, x);
    // FF1 + gelu (strip 256 -> A read 2x)
    gemm_bf16_kernel<16><<<dim3(2, TOK / 128), 256, 0, stream>>>(
        x, w1_bf + (size_t)l * FFD * DD, b1 + l * FFD, big, FFD, DD, 1);
    // FF2
    gemm_bf16_kernel<8><<<dim3(1, TOK / 128), 256, 0, stream>>>(
        big, w2_bf + (size_t)l * DD * FFD, b2 + l * DD, t2, DD, FFD, 0);
    ln_kernel<<<TOK, 128, 0, stream>>>(x, t2, ln2g + l * DD, ln2b + l * DD, x);
  }

  mean_kernel<<<BSZ, 128, 0, stream>>>(x, gemb);
  // keys = node_emb @ Wk^T
  gemm_bf16_kernel<8><<<dim3(1, TOK / 128), 256, 0, stream>>>(
      x, wk_bf, nullptr, keys, DD, DD, 0);

  decode_kernel<<<BSZ, 256, DEC_LDS, stream>>>(x, keys, gemb, Wq, out);
}
